// MultiHeadSelfAttention_59150289600583
// MI455X (gfx1250) — matrix operand order repaired and checked by reading, not compiled, not run on hardware
//
#include <hip/hip_runtime.h>
#include <hip/hip_bf16.h>

typedef __attribute__((ext_vector_type(16))) _Float16 v16h;
typedef __attribute__((ext_vector_type(8)))  _Float16 v8h;
typedef __attribute__((ext_vector_type(8)))  float    v8f;

#define EMBED  1024
#define T_SEQ  2048
#define NHEAD  16
#define HEAD   64
#define NROWS  4096   /* B*T */
#define FFDIM  4096
#define LDSLD  40     /* 32 halves + 8 pad -> conflict-free b128 LDS reads */
#define LBUF   (128 * LDSLD)   /* one 128x32 chunk (halves) */

// ---------------------------------------------------------------------------
// WMMA fragment helpers (wave32, gfx1250 layouts from CDNA5 ISA 7.12.2)
// ---------------------------------------------------------------------------
__device__ __forceinline__ v16h make_frag(const _Float16* p0, const _Float16* p1) {
  v8h lo = *(const v8h*)p0;
  v8h hi = *(const v8h*)p1;
  v16h a;
#pragma unroll
  for (int j = 0; j < 8; ++j) { a[j] = lo[j]; a[j + 8] = hi[j]; }
  return a;
}

// A fragment 16x32 f16 from row-major [.. x ld] at (rowbase, kk).
// lane L: row = L%16, k chunks at (L/16)*8 and 16+(L/16)*8 (each 8 halves).
__device__ __forceinline__ v16h a_frag(const _Float16* base, int ld, int rowbase,
                                       int kk, int lane) {
  const int m = lane & 15, half = lane >> 4;
  const _Float16* p = base + (size_t)(rowbase + m) * ld + kk + half * 8;
  return make_frag(p, p + 16);
}

// B fragment 32x16 f16 from B-transposed row-major Bt[n][k] at (colbase, kk).
// lane L: col n = L%16, k chunks at (L/16)*8 and 16+(L/16)*8 (each 8 halves).
__device__ __forceinline__ v16h b_frag(const _Float16* baseT, int ld, int colbase,
                                       int kk, int lane) {
  const int n = lane & 15, g = lane >> 4;
  const _Float16* p = baseT + (size_t)(colbase + n) * ld + kk + g * 8;
  return make_frag(p, p + 16);
}

__device__ __forceinline__ v8f wmma_f16(v16h a, v16h b, v8f c) {
  return __builtin_amdgcn_wmma_f32_16x16x32_f16(false, a, false, b, (short)0, c,
                                                false, false);
}

// ---------------------------------------------------------------------------
// CDNA5 async global->LDS copy (ASYNCcnt path, ISA §10.4): per-lane 16B DMA,
// no VGPR bounce buffer, no loadcnt stall in the staging path.
// VDST = wave-relative LDS byte offset (low 32 bits of generic shared ptr).
// ---------------------------------------------------------------------------
__device__ __forceinline__ void async_ld_b128(_Float16* lptr, const _Float16* gptr) {
  unsigned loff = (unsigned)(unsigned long long)lptr;
  asm volatile("global_load_async_to_lds_b128 %0, %1, off"
               :
               : "v"(loff), "v"(gptr)
               : "memory");
}

__device__ __forceinline__ void wait_async() {
#if __has_builtin(__builtin_amdgcn_s_wait_asynccnt)
  __builtin_amdgcn_s_wait_asynccnt(0);
#else
  asm volatile("s_wait_asynccnt 0x0" ::: "memory");
#endif
}

// ---------------------------------------------------------------------------
// Weight conversion kernels (fp32 -> f16, transposed for B-frag loads)
// ---------------------------------------------------------------------------
__global__ void cvt_wqkv_kernel(const float* __restrict__ wq, const float* __restrict__ wk,
                                const float* __restrict__ wv, const float* __restrict__ bq,
                                const float* __restrict__ bk, const float* __restrict__ bv,
                                _Float16* __restrict__ wqkvT, float* __restrict__ qkvbias) {
  int id = blockIdx.x * blockDim.x + threadIdx.x;      // over 3072*1024
  int n = id >> 10, c = id & 1023;
  int p = n >> 10, rem = n & 1023, h = rem >> 6, d = rem & 63;
  const float* W = (p == 0) ? wq : (p == 1) ? wk : wv; // [NHEAD][EMBED][HEAD]
  wqkvT[id] = (_Float16)W[((size_t)h * EMBED + c) * HEAD + d];
  if (c == 0) {
    const float* B = (p == 0) ? bq : (p == 1) ? bk : bv;
    qkvbias[n] = B[h * HEAD + d];
  }
}

// dst[c][r] = src[r][c]   (src fp32 [R][C] -> dst f16 [C][R])
__global__ void transpose_cvt_kernel(const float* __restrict__ src,
                                     _Float16* __restrict__ dst, int R, int C) {
  int id = blockIdx.x * blockDim.x + threadIdx.x;      // over R*C, r fastest
  int r = id % R, c = id / R;
  dst[(size_t)c * R + r] = (_Float16)src[(size_t)r * C + c];
}

// ---------------------------------------------------------------------------
// LayerNorm (optionally fused residual add), writes f32 + f16 copies
// ---------------------------------------------------------------------------
__global__ void __launch_bounds__(256)
ln_kernel(const float* __restrict__ X, const float* __restrict__ add, int use_add,
          const float* __restrict__ g, const float* __restrict__ bb,
          float* __restrict__ out32, _Float16* __restrict__ out16) {
  const int row = blockIdx.x;
  const size_t off = (size_t)row * EMBED;
  const int tid = threadIdx.x;
  float s = 0.f, ss = 0.f, vals[4];
#pragma unroll
  for (int i = 0; i < 4; ++i) {
    int c = tid + i * 256;
    float v = X[off + c];
    if (use_add) v += add[off + c];
    vals[i] = v; s += v; ss += v * v;
  }
#pragma unroll
  for (int d = 16; d > 0; d >>= 1) { s += __shfl_xor(s, d); ss += __shfl_xor(ss, d); }
  __shared__ float rs[8], rss[8], s_mu, s_rstd;
  const int lane = tid & 31, wave = tid >> 5;
  if (lane == 0) { rs[wave] = s; rss[wave] = ss; }
  __syncthreads();
  if (tid == 0) {
    float S = 0.f, SS = 0.f;
    for (int i = 0; i < 8; ++i) { S += rs[i]; SS += rss[i]; }
    float mu = S * (1.0f / EMBED);
    float var = SS * (1.0f / EMBED) - mu * mu;
    s_mu = mu; s_rstd = rsqrtf(var + 1e-5f);
  }
  __syncthreads();
  const float mu = s_mu, rstd = s_rstd;
#pragma unroll
  for (int i = 0; i < 4; ++i) {
    int c = tid + i * 256;
    float o = (vals[i] - mu) * rstd * g[c] + bb[c];
    out32[off + c] = o;
    out16[off + c] = (_Float16)o;
  }
}

// ---------------------------------------------------------------------------
// Shared WMMA GEMM mainloop: 128x128 block tile, 8 waves (4M x 2N), each wave
// computes 32x64 (2x4 D-tiles). Double-buffered LDS with ASYNC global->LDS
// staging: chunk kk+32 DMAs into the other buffer while WMMAs consume chunk
// kk; s_wait_asynccnt drains only at the barrier. Speculative prefetch ahead.
// ---------------------------------------------------------------------------
__device__ __forceinline__ void stage_chunk(const _Float16* __restrict__ A,
                                            const _Float16* __restrict__ Bt, int K,
                                            int mbase, int nbase, int kk,
                                            _Float16* lA, _Float16* lB, int tid) {
#pragma unroll
  for (int rpt = 0; rpt < 2; ++rpt) {
    int idx = tid + rpt * 256;
    int row = idx >> 2, seg = idx & 3;
    const _Float16* ga = A + (size_t)(mbase + row) * K + kk + seg * 8;
    const _Float16* gb = Bt + (size_t)(nbase + row) * K + kk + seg * 8;
    async_ld_b128(lA + row * LDSLD + seg * 8, ga);
    async_ld_b128(lB + row * LDSLD + seg * 8, gb);
    // speculative prefetch of the chunk after next (safe past end: dropped)
    __builtin_prefetch(ga + 64, 0, 3);
    __builtin_prefetch(gb + 64, 0, 3);
  }
}

__device__ __forceinline__ void gemm_mainloop(const _Float16* __restrict__ A,
                                              const _Float16* __restrict__ Bt, int K,
                                              int mbase, int nbase, v8f acc[2][4],
                                              _Float16* lA, _Float16* lB) {
  const int tid = threadIdx.x;
  const int lane = tid & 31;
  const int wave = tid >> 5;
  const int wm = wave & 3, wn = wave >> 2;

  stage_chunk(A, Bt, K, mbase, nbase, 0, lA, lB, tid);
  wait_async();
  __syncthreads();

  for (int kk = 0; kk < K; kk += 32) {
    const int cur = (kk >> 5) & 1;
    _Float16* cA = lA + cur * LBUF;
    _Float16* cB = lB + cur * LBUF;
    if (kk + 32 < K)   // async-stage next chunk into the other buffer
      stage_chunk(A, Bt, K, mbase, nbase, kk + 32,
                  lA + (1 - cur) * LBUF, lB + (1 - cur) * LBUF, tid);
    v16h af[2], bf[4];
#pragma unroll
    for (int mt = 0; mt < 2; ++mt) af[mt] = a_frag(cA, LDSLD, wm * 32 + mt * 16, 0, lane);
#pragma unroll
    for (int nt = 0; nt < 4; ++nt) bf[nt] = b_frag(cB, LDSLD, wn * 64 + nt * 16, 0, lane);
#pragma unroll
    for (int mt = 0; mt < 2; ++mt)
#pragma unroll
      for (int nt = 0; nt < 4; ++nt)
        acc[mt][nt] = wmma_f16(af[mt], bf[nt], acc[mt][nt]);
    wait_async();     // drain this wave's async DMAs, then rendezvous
    __syncthreads();
  }
}

// QKV GEMM: h[4096x1024] x WqkvT -> scatter q[t][d], k[t][d], vT[d][t] per head
__global__ void __launch_bounds__(256)
gemm_qkv_kernel(const _Float16* __restrict__ A, const _Float16* __restrict__ Bt,
                const float* __restrict__ bias, _Float16* __restrict__ qf,
                _Float16* __restrict__ kf, _Float16* __restrict__ vtf) {
  __shared__ _Float16 lA[2 * LBUF];
  __shared__ _Float16 lB[2 * LBUF];
  const int mbase = blockIdx.y * 128, nbase = blockIdx.x * 128;
  v8f acc[2][4];
  v8f zero = {};
#pragma unroll
  for (int mt = 0; mt < 2; ++mt)
#pragma unroll
    for (int nt = 0; nt < 4; ++nt) acc[mt][nt] = zero;
  gemm_mainloop(A, Bt, EMBED, mbase, nbase, acc, lA, lB);

  const int lane = threadIdx.x & 31, wave = threadIdx.x >> 5;
  const int wm = wave & 3, wn = wave >> 2;
  const int n16 = lane & 15, half = lane >> 4;
#pragma unroll
  for (int mt = 0; mt < 2; ++mt)
#pragma unroll
    for (int nt = 0; nt < 4; ++nt) {
      int colg = nbase + wn * 64 + nt * 16 + n16;       // [0,3072)
      int p = colg >> 10, rem = colg & 1023, h = rem >> 6, d = rem & 63;
      float bv = bias[colg];
#pragma unroll
      for (int r = 0; r < 8; ++r) {
        int rowg = mbase + wm * 32 + mt * 16 + r + 8 * half;  // [0,4096)
        int bidx = rowg >> 11, t = rowg & 2047;
        int bh = bidx * NHEAD + h;
        float val = acc[mt][nt][r] + bv;
        if (p == 0)      qf[((size_t)bh * T_SEQ + t) * HEAD + d] = (_Float16)val;
        else if (p == 1) kf[((size_t)bh * T_SEQ + t) * HEAD + d] = (_Float16)val;
        else             vtf[((size_t)bh * HEAD + d) * T_SEQ + t] = (_Float16)val;
      }
    }
}

// FFN1: z[4096x1024] x W1t -> relu(+b1), f16 out [4096x4096]
__global__ void __launch_bounds__(256)
gemm_ffn1_kernel(const _Float16* __restrict__ A, const _Float16* __restrict__ Bt,
                 const float* __restrict__ bias, _Float16* __restrict__ out) {
  __shared__ _Float16 lA[2 * LBUF];
  __shared__ _Float16 lB[2 * LBUF];
  const int mbase = blockIdx.y * 128, nbase = blockIdx.x * 128;
  v8f acc[2][4];
  v8f zero = {};
#pragma unroll
  for (int mt = 0; mt < 2; ++mt)
#pragma unroll
    for (int nt = 0; nt < 4; ++nt) acc[mt][nt] = zero;
  gemm_mainloop(A, Bt, EMBED, mbase, nbase, acc, lA, lB);

  const int lane = threadIdx.x & 31, wave = threadIdx.x >> 5;
  const int wm = wave & 3, wn = wave >> 2;
  const int n16 = lane & 15, half = lane >> 4;
#pragma unroll
  for (int mt = 0; mt < 2; ++mt)
#pragma unroll
    for (int nt = 0; nt < 4; ++nt) {
      int colg = nbase + wn * 64 + nt * 16 + n16;
      float bv = bias[colg];
#pragma unroll
      for (int r = 0; r < 8; ++r) {
        int rowg = mbase + wm * 32 + mt * 16 + r + 8 * half;
        out[(size_t)rowg * FFDIM + colg] = (_Float16)fmaxf(acc[mt][nt][r] + bv, 0.f);
      }
    }
}

// FFN2: act[4096x4096] x W2t -> + b2 + residual z, fp32 out
__global__ void __launch_bounds__(256)
gemm_ffn2_kernel(const _Float16* __restrict__ A, const _Float16* __restrict__ Bt,
                 const float* __restrict__ bias, const float* __restrict__ resid,
                 float* __restrict__ out) {
  __shared__ _Float16 lA[2 * LBUF];
  __shared__ _Float16 lB[2 * LBUF];
  const int mbase = blockIdx.y * 128, nbase = blockIdx.x * 128;
  v8f acc[2][4];
  v8f zero = {};
#pragma unroll
  for (int mt = 0; mt < 2; ++mt)
#pragma unroll
    for (int nt = 0; nt < 4; ++nt) acc[mt][nt] = zero;
  gemm_mainloop(A, Bt, FFDIM, mbase, nbase, acc, lA, lB);

  const int lane = threadIdx.x & 31, wave = threadIdx.x >> 5;
  const int wm = wave & 3, wn = wave >> 2;
  const int n16 = lane & 15, half = lane >> 4;
#pragma unroll
  for (int mt = 0; mt < 2; ++mt)
#pragma unroll
    for (int nt = 0; nt < 4; ++nt) {
      int colg = nbase + wn * 64 + nt * 16 + n16;
      float bv = bias[colg];
#pragma unroll
      for (int r = 0; r < 8; ++r) {
        int rowg = mbase + wm * 32 + mt * 16 + r + 8 * half;
        size_t o = (size_t)rowg * EMBED + colg;
        out[o] = resid[o] + acc[mt][nt][r] + bv;
      }
    }
}

// ---------------------------------------------------------------------------
// Flash attention: one wave per (bh, 16-query tile). Computes S^T = K.Q^T so
// the D-layout of scores (lane=query, vgprs=keys) IS the A-layout of P for
// the P.V WMMA when consuming 32 keys/chunk -> zero cross-lane transposes.
// ---------------------------------------------------------------------------
__global__ void __launch_bounds__(128)
attn_kernel(const _Float16* __restrict__ qf, const _Float16* __restrict__ kf,
            const _Float16* __restrict__ vtf, float* __restrict__ attn_out) {
  const int lane = threadIdx.x & 31;
  const int wave = threadIdx.x >> 5;
  const int bh = blockIdx.y;                 // 0..31
  const int qtile = blockIdx.x * 4 + wave;   // 0..127
  const int qbase = qtile * 16;
  const int n = lane & 15, half = lane >> 4;

  const _Float16* Q  = qf + (size_t)bh * T_SEQ * HEAD;
  const _Float16* Km = kf + (size_t)bh * T_SEQ * HEAD;
  const _Float16* Vt = vtf + (size_t)bh * HEAD * T_SEQ;

  // Q as B-matrix (B^t = row-major q): reused across every key chunk
  const v16h qb0 = b_frag(Q, HEAD, qbase, 0, lane);
  const v16h qb1 = b_frag(Q, HEAD, qbase, 32, lane);

  v8f zero = {};
  v8f o[4];
#pragma unroll
  for (int t = 0; t < 4; ++t) o[t] = zero;
  float m2 = -1e30f, lsum = 0.f;
  const int qg = qbase + n;                  // this lane's query index
  const float SC = 0.125f * 1.44269504f;     // 1/sqrt(64) * log2(e)

  for (int s = 0; s < qbase + 16; s += 32) {
    // S^T tiles: rows = keys, cols = queries  (2 wmma each, K-dim = 64)
    v8f sa = zero, sb = zero;
    sa = wmma_f16(a_frag(Km, HEAD, s, 0, lane), qb0, sa);
    sa = wmma_f16(a_frag(Km, HEAD, s, 32, lane), qb1, sa);
    sb = wmma_f16(a_frag(Km, HEAD, s + 16, 0, lane), qb0, sb);
    sb = wmma_f16(a_frag(Km, HEAD, s + 16, 32, lane), qb1, sb);

    float va[8], vb[8], mx = -1e30f;
#pragma unroll
    for (int r = 0; r < 8; ++r) {
      int key = s + r + 8 * half;
      va[r] = (key <= qg) ? sa[r] * SC : -1e30f;          // causal mask
      vb[r] = (key + 16 <= qg) ? sb[r] * SC : -1e30f;
      mx = fmaxf(mx, fmaxf(va[r], vb[r]));
    }
    mx = fmaxf(mx, __shfl_xor(mx, 16));                    // max over 32 keys
    float m2n = fmaxf(m2, mx);
    float alpha = exp2f(m2 - m2n);
    m2 = m2n;

    v16h pf;                                               // P A-fragment
    float rs = 0.f;
#pragma unroll
    for (int r = 0; r < 8; ++r) {
      float pa = exp2f(va[r] - m2n);
      float pb = exp2f(vb[r] - m2n);
      rs += pa + pb;
      pf[r] = (_Float16)pa;                                // keys half*8+r
      pf[r + 8] = (_Float16)pb;                            // keys 16+half*8+r
    }
    rs += __shfl_xor(rs, 16);
    lsum = lsum * alpha + rs;

    float ar[8];                                           // alpha per O-row
#pragma unroll
    for (int r = 0; r < 8; ++r) ar[r] = __shfl(alpha, r + 8 * half);

#pragma unroll
    for (int t = 0; t < 4; ++t) {
      v8f c = o[t];
#pragma unroll
      for (int r = 0; r < 8; ++r) c[r] *= ar[r];
      // V as B-matrix via transposed vT[d][t] store
      o[t] = wmma_f16(pf, b_frag(Vt, T_SEQ, t * 16, s, lane), c);
    }
  }

  float lr[8];
#pragma unroll
  for (int r = 0; r < 8; ++r) lr[r] = __shfl(lsum, r + 8 * half);
  const int b = bh >> 4, h = bh & 15;
#pragma unroll
  for (int t = 0; t < 4; ++t)
#pragma unroll
    for (int r = 0; r < 8; ++r) {
      int row = b * T_SEQ + qbase + r + 8 * half;
      int col = h * HEAD + t * 16 + n;
      attn_out[(size_t)row * EMBED + col] = o[t][r] / lr[r];
    }
}

// ---------------------------------------------------------------------------
extern "C" void kernel_launch(void* const* d_in, const int* in_sizes, int n_in,
                              void* d_out, int out_size, void* d_ws, size_t ws_size,
                              hipStream_t stream) {
  const float* x     = (const float*)d_in[0];
  const float* wq    = (const float*)d_in[1];
  const float* bq    = (const float*)d_in[2];
  const float* wk    = (const float*)d_in[3];
  const float* bk    = (const float*)d_in[4];
  const float* wv    = (const float*)d_in[5];
  const float* bv    = (const float*)d_in[6];
  const float* ln1_g = (const float*)d_in[7];
  const float* ln1_b = (const float*)d_in[8];
  const float* ln2_g = (const float*)d_in[9];
  const float* ln2_b = (const float*)d_in[10];
  const float* w1    = (const float*)d_in[11];
  const float* b1    = (const float*)d_in[12];
  const float* w2    = (const float*)d_in[13];
  const float* b2    = (const float*)d_in[14];
  float* out = (float*)d_out;

  char* wp = (char*)d_ws;
  auto carve = [&](size_t bytes) {
    char* p = wp;
    wp += (bytes + 255) & ~(size_t)255;
    return (void*)p;
  };
  float*    hf32    = (float*)   carve((size_t)NROWS * EMBED * 4);
  _Float16* hf16    = (_Float16*)carve((size_t)NROWS * EMBED * 2);
  _Float16* wqkvT   = (_Float16*)carve((size_t)3 * EMBED * EMBED * 2);
  float*    qkvbias = (float*)   carve((size_t)3 * EMBED * 4);
  _Float16* qf      = (_Float16*)carve((size_t)32 * T_SEQ * HEAD * 2);
  _Float16* kf      = (_Float16*)carve((size_t)32 * T_SEQ * HEAD * 2);
  _Float16* vtf     = (_Float16*)carve((size_t)32 * HEAD * T_SEQ * 2);
  float*    attnb   = (float*)   carve((size_t)NROWS * EMBED * 4);
  float*    zf32    = (float*)   carve((size_t)NROWS * EMBED * 4);
  _Float16* zf16    = (_Float16*)carve((size_t)NROWS * EMBED * 2);
  _Float16* w1t     = (_Float16*)carve((size_t)EMBED * FFDIM * 2);
  _Float16* w2t     = (_Float16*)carve((size_t)FFDIM * EMBED * 2);
  _Float16* actf    = (_Float16*)carve((size_t)NROWS * FFDIM * 2);

  // 1) weight conversion
  cvt_wqkv_kernel<<<(3 * EMBED * EMBED) / 256, 256, 0, stream>>>(
      wq, wk, wv, bq, bk, bv, wqkvT, qkvbias);
  transpose_cvt_kernel<<<(EMBED * FFDIM) / 256, 256, 0, stream>>>(w1, w1t, EMBED, FFDIM);
  transpose_cvt_kernel<<<(FFDIM * EMBED) / 256, 256, 0, stream>>>(w2, w2t, FFDIM, EMBED);

  // 2) LN1: h = LN(x)
  ln_kernel<<<NROWS, 256, 0, stream>>>(x, x, 0, ln1_g, ln1_b, hf32, hf16);

  // 3) QKV projection (M=4096, N=3072, K=1024)
  gemm_qkv_kernel<<<dim3(3 * EMBED / 128, NROWS / 128), 256, 0, stream>>>(
      hf16, wqkvT, qkvbias, qf, kf, vtf);

  // 4) causal flash attention
  attn_kernel<<<dim3(T_SEQ / 16 / 4, 32), 128, 0, stream>>>(qf, kf, vtf, attnb);

  // 5) LN2: z = LN(h + attn)
  ln_kernel<<<NROWS, 256, 0, stream>>>(hf32, attnb, 1, ln2_g, ln2_b, zf32, zf16);

  // 6) FFN1: relu(z @ w1 + b1)  (M=4096, N=4096, K=1024)
  gemm_ffn1_kernel<<<dim3(FFDIM / 128, NROWS / 128), 256, 0, stream>>>(
      zf16, w1t, b1, actf);

  // 7) FFN2: out = z + act @ w2 + b2  (M=4096, N=1024, K=4096)
  gemm_ffn2_kernel<<<dim3(EMBED / 128, NROWS / 128), 256, 0, stream>>>(
      actf, w2t, b2, zf32, out);
}
